// BeamSearchAdapter_5274219839721
// MI455X (gfx1250) — matrix operand order repaired and checked by reading, not compiled, not run on hardware
//
#include <hip/hip_runtime.h>
#include <hip/hip_bf16.h>
#include <math.h>

// ---------------------------------------------------------------------------
// Beam-search decoder for MI455X (gfx1250).
// All matmuls are M=64-row skinny GEMMs -> HBM-bandwidth bound, so weights are
// converted once per launch to bf16 *transposed* layout, letting every WMMA
// operand (A and B) be fetched as two 16-byte vector loads per 32-K chunk.
// Matrix math: v_wmma_f32_16x16x32_bf16, fp32 accumulate/states.
// ---------------------------------------------------------------------------

typedef __bf16 bf16_t;
typedef __attribute__((ext_vector_type(16))) __bf16 v16bf;
typedef __attribute__((ext_vector_type(8)))  float  v8f;

#define BB 8          // batch
#define KB 8          // beam width
#define RR 64         // rows = B*K
#define TT 32         // time steps
#define LL 256        // encoder length
#define HH 1024       // hidden
#define DE 512        // embedding dim
#define VV 32000      // vocab
#define EOS_TOK 2
#define NEG_BIG (-3.0e38f)

__device__ __forceinline__ float sigf(float x) { return 1.0f / (1.0f + __expf(-x)); }

// ---------------------------------------------------------------------------
// fp32 -> bf16 (grid-stride), plain layout
// ---------------------------------------------------------------------------
__global__ void f2bf_kernel(const float* __restrict__ s, bf16_t* __restrict__ d, long n) {
  long i = (long)blockIdx.x * blockDim.x + threadIdx.x;
  long stride = (long)gridDim.x * blockDim.x;
  for (; i < n; i += stride) d[i] = (bf16_t)s[i];
}

// ---------------------------------------------------------------------------
// fp32 [R x C] -> bf16 [C x R] tiled transpose (coalesced read & write, LDS).
// R, C multiples of 32. gridDim.z = batch with given strides.
// ---------------------------------------------------------------------------
__global__ void f2bfT_kernel(const float* __restrict__ src, bf16_t* __restrict__ dst,
                             int R, int C, long sStride, long dStride) {
  const float* s = src + (long)blockIdx.z * sStride;
  bf16_t* d = dst + (long)blockIdx.z * dStride;
  __shared__ bf16_t tile[32][33];
  const int c0 = blockIdx.x * 32, r0 = blockIdx.y * 32;
  const int tx = threadIdx.x & 31, ty = threadIdx.x >> 5;   // 256 thr = 32x8
#pragma unroll
  for (int rr = ty; rr < 32; rr += 8)
    tile[rr][tx] = (bf16_t)s[(long)(r0 + rr) * C + c0 + tx];
  __syncthreads();
#pragma unroll
  for (int cc = ty; cc < 32; cc += 8)
    d[(long)(c0 + cc) * R + r0 + tx] = tile[tx][cc];
}

// ---------------------------------------------------------------------------
// One 16x16 WMMA output tile: acc += A(16 x K) * B(K x 16), bf16 in, f32 out.
// A  : row-major, lda elements (16B-aligned rows; lda % 8 == 0).
// Bt : column-major-per-output-column: element (k, n) at Bt + n*ldb + k.
// Both fragments load as two uint4 (16B) per chunk -- no scalar u16 gathers.
// Lane layouts per CDNA5 ISA 7.12.2 (wave32).
// ---------------------------------------------------------------------------
__device__ __forceinline__ v8f wmma_tile(const bf16_t* __restrict__ A, int lda,
                                         const bf16_t* __restrict__ Bt, int ldb,
                                         int K, int mlimit) {
  const int lane = threadIdx.x & 31;
  const int half = lane >> 4;   // 0: lanes 0-15, 1: lanes 16-31
  const int l15  = lane & 15;
  const int arow = (l15 < mlimit) ? l15 : (mlimit - 1);  // clamp padded rows
  const bf16_t* Ar = A + (long)arow * lda;
  const bf16_t* Bc = Bt + (long)l15 * ldb;
  v8f acc = {};
  for (int k0 = 0; k0 < K; k0 += 32) {
    // Stream-prefetch the B column a few chunks ahead (global_prefetch_b8).
    __builtin_prefetch((const void*)(Bc + k0 + 128), 0, 0);
    union { v16bf v; uint4 q[2]; } a, b;
    // A 16x32 layout: e=0..7 -> k0+8*half+e ; e=8..15 -> k0+16+8*half+(e-8)
    a.q[0] = *(const uint4*)(Ar + k0 + 8 * half);
    a.q[1] = *(const uint4*)(Ar + k0 + 16 + 8 * half);
    // B 32x16 layout: lane = column, e=0..7 -> k0+8*half+e ; e=8..15 -> k0+16+8*half+(e-8)
    b.q[0] = *(const uint4*)(Bc + k0 + 8 * half);
    b.q[1] = *(const uint4*)(Bc + k0 + 16 + 8 * half);
    acc = __builtin_amdgcn_wmma_f32_16x16x32_bf16(
        /*neg_a=*/false, a.v, /*neg_b=*/false, b.v,
        /*c_mod=*/(short)0, acc, /*reuse_a=*/false, /*reuse_b=*/false);
  }
  return acc;
}

// ---------------------------------------------------------------------------
// Generic GEMM: C[M x N] = A[M x K](bf16) * W(K x N) + bias, fp32 out.
// W supplied transposed: Wt[N x K]. grid.y = M/16, each wave owns one N tile.
// ---------------------------------------------------------------------------
__global__ void gemm_kernel(const bf16_t* __restrict__ A, int lda,
                            const bf16_t* __restrict__ Wt, int ldb,
                            const float* __restrict__ bias,
                            float* __restrict__ C, int ldc, int N, int K) {
  const int w = threadIdx.x >> 5;
  const int ntile = blockIdx.x * (blockDim.x >> 5) + w;
  const int mtile = blockIdx.y;
  if (ntile * 16 >= N) return;                       // wave-uniform
  v8f acc = wmma_tile(A + (long)mtile * 16 * lda, lda,
                      Wt + (long)ntile * 16 * ldb, ldb, K, 16);
  const int lane = threadIdx.x & 31, half = lane >> 4, n = lane & 15;
  const float bv = bias ? bias[ntile * 16 + n] : 0.0f;
  float* Ct = C + (long)mtile * 16 * ldc + ntile * 16 + n;
#pragma unroll
  for (int r = 0; r < 8; ++r)
    Ct[(long)(r + 8 * half) * ldc] = acc[r] + bv;
}

// scores[b,k,l] = sum_e h2[b,k,e] * enc[b,l,e]
// "column l" of the product reads enc row l -> K-contiguous: pass encb rows.
__global__ void scores_kernel(const bf16_t* __restrict__ h2b,
                              const bf16_t* __restrict__ encb,
                              float* __restrict__ scores) {
  const int w = threadIdx.x >> 5;
  const int ltile = blockIdx.x * (blockDim.x >> 5) + w;
  const int b = blockIdx.y;
  if (ltile * 16 >= LL) return;
  v8f acc = wmma_tile(h2b + (long)b * KB * HH, HH,
                      encb + ((long)b * LL + ltile * 16) * HH, HH, HH, KB);
  const int lane = threadIdx.x & 31, half = lane >> 4, n = lane & 15;
#pragma unroll
  for (int r = 0; r < 8; ++r) {
    int m = r + 8 * half;
    if (m < KB) scores[((long)(b * KB + m)) * LL + ltile * 16 + n] = acc[r];
  }
}

// ctx[b,k,h] = sum_l att[b,k,l] * enc[b,l,h]; uses encT[b][h][l] for
// K-contiguous column loads. Stores bf16 into XC second half.
__global__ void ctx_kernel(const bf16_t* __restrict__ attb,
                           const bf16_t* __restrict__ encTb,
                           bf16_t* __restrict__ XC) {
  const int w = threadIdx.x >> 5;
  const int ntile = blockIdx.x * (blockDim.x >> 5) + w;
  const int b = blockIdx.y;
  if (ntile * 16 >= HH) return;
  v8f acc = wmma_tile(attb + (long)b * KB * LL, LL,
                      encTb + (long)b * HH * LL + (long)ntile * 16 * LL, LL, LL, KB);
  const int lane = threadIdx.x & 31, half = lane >> 4, n = lane & 15;
#pragma unroll
  for (int r = 0; r < 8; ++r) {
    int m = r + 8 * half;
    if (m < KB)
      XC[((long)(b * KB + m)) * 2048 + HH + ntile * 16 + n] = (bf16_t)acc[r];
  }
}

// ---------------------------------------------------------------------------
// State init: broadcast [B,H] initial states to [B,K,H]; hatt0 = c2.
// State layout S[5][64][1024] = {hatt, h1, c1, h2, c2}
// ---------------------------------------------------------------------------
__global__ void init_kernel(const int* __restrict__ summ,
                            const float* __restrict__ h1, const float* __restrict__ c1,
                            const float* __restrict__ h2, const float* __restrict__ c2,
                            float* __restrict__ S, int* __restrict__ tok,
                            int* __restrict__ mask, float* __restrict__ nls) {
  int i = blockIdx.x * blockDim.x + threadIdx.x;    // 64*1024 threads
  int row = i >> 10, j = i & (HH - 1);
  int b = row >> 3, k = row & 7;
  S[0 * RR * HH + row * HH + j] = c2[b * HH + j];   // hatt0 = c2
  S[1 * RR * HH + row * HH + j] = h1[b * HH + j];
  S[2 * RR * HH + row * HH + j] = c1[b * HH + j];
  S[3 * RR * HH + row * HH + j] = h2[b * HH + j];
  S[4 * RR * HH + row * HH + j] = c2[b * HH + j];
  if (j == 0) {
    tok[row]  = summ[b * TT + 0];
    mask[row] = 1;
    nls[row]  = (k == 0) ? 0.0f : -1e30f;           // virtualize K=1 first step
  }
}

__global__ void mask_kernel(const int* __restrict__ tok, int* __restrict__ mask) {
  int r = threadIdx.x;
  if (r < RR) mask[r] = mask[r] && (tok[r] != EOS_TOK);
}

// X1 = [emb[tok] | hatt | h1] in bf16, [64 x 2560]
__global__ void build_x1_kernel(const int* __restrict__ tok, const float* __restrict__ emb,
                                const float* __restrict__ S, bf16_t* __restrict__ X1) {
  int i = blockIdx.x * blockDim.x + threadIdx.x;    // 64*2560
  int row = i / 2560, c = i % 2560;
  float v;
  if (c < DE)            v = emb[(long)tok[row] * DE + c];
  else if (c < DE + HH)  v = S[0 * RR * HH + row * HH + (c - DE)];        // hatt
  else                   v = S[1 * RR * HH + row * HH + (c - DE - HH)];   // h1
  X1[(long)row * 2560 + c] = (bf16_t)v;
}

// LSTM1 pointwise: update h1,c1; X2 = [h1_new | h2_prev] bf16
__global__ void lstm1_pw_kernel(const float* __restrict__ Z, float* __restrict__ S,
                                bf16_t* __restrict__ X2) {
  int i = blockIdx.x * blockDim.x + threadIdx.x;    // 64*1024
  int row = i >> 10, j = i & (HH - 1);
  const float* z = Z + (long)row * 4 * HH;
  float iv = sigf(z[j]);
  float fv = sigf(z[HH + j]);
  float gv = tanhf(z[2 * HH + j]);
  float ov = sigf(z[3 * HH + j]);
  float cn = fv * S[2 * RR * HH + row * HH + j] + iv * gv;
  float hn = ov * tanhf(cn);
  S[2 * RR * HH + row * HH + j] = cn;
  S[1 * RR * HH + row * HH + j] = hn;
  X2[(long)row * 2048 + j]      = (bf16_t)hn;
  X2[(long)row * 2048 + HH + j] = (bf16_t)S[3 * RR * HH + row * HH + j];  // h2 prev
}

// LSTM2 pointwise: update h2,c2; stage h2 bf16 + XC first half
__global__ void lstm2_pw_kernel(const float* __restrict__ Z, float* __restrict__ S,
                                bf16_t* __restrict__ h2b, bf16_t* __restrict__ XC) {
  int i = blockIdx.x * blockDim.x + threadIdx.x;
  int row = i >> 10, j = i & (HH - 1);
  const float* z = Z + (long)row * 4 * HH;
  float iv = sigf(z[j]);
  float fv = sigf(z[HH + j]);
  float gv = tanhf(z[2 * HH + j]);
  float ov = sigf(z[3 * HH + j]);
  float cn = fv * S[4 * RR * HH + row * HH + j] + iv * gv;
  float hn = ov * tanhf(cn);
  S[4 * RR * HH + row * HH + j] = cn;
  S[3 * RR * HH + row * HH + j] = hn;
  bf16_t hb = (bf16_t)hn;
  h2b[(long)row * HH + j]  = hb;
  XC[(long)row * 2048 + j] = hb;
}

// softmax over L=256 per row; emit bf16 attention weights
__global__ void softmax_kernel(const float* __restrict__ scores, bf16_t* __restrict__ attb) {
  int row = blockIdx.x, tid = threadIdx.x;          // 256 threads, 256 cols
  __shared__ float red[256];
  float x = scores[(long)row * LL + tid];
  red[tid] = x; __syncthreads();
  for (int s = 128; s > 0; s >>= 1) { if (tid < s) red[tid] = fmaxf(red[tid], red[tid + s]); __syncthreads(); }
  float mx = red[0]; __syncthreads();
  float e = __expf(x - mx);
  red[tid] = e; __syncthreads();
  for (int s = 128; s > 0; s >>= 1) { if (tid < s) red[tid] += red[tid + s]; __syncthreads(); }
  attb[(long)row * LL + tid] = (bf16_t)(e / red[0]);
}

// hatt = tanh(pre); stage bf16 copy for vocab GEMM
__global__ void tanh_pw_kernel(const float* __restrict__ Hpre, float* __restrict__ S,
                               bf16_t* __restrict__ hattb) {
  int i = blockIdx.x * blockDim.x + threadIdx.x;
  int row = i >> 10, j = i & (HH - 1);
  float v = tanhf(Hpre[(long)row * HH + j]);
  S[0 * RR * HH + row * HH + j] = v;
  hattb[(long)row * HH + j] = (bf16_t)v;
}

// Per-row: logsumexp + top-8 of logits; apply mask + accumulated score.
// cand = mask ? (logit - lse) + nlogsum : nlogsum ; idx = mask ? idx : 0
__global__ void topk_kernel(const float* __restrict__ logits,
                            const int* __restrict__ mask, const float* __restrict__ nls,
                            float* __restrict__ cvals, int* __restrict__ cidx) {
  const int row = blockIdx.x, tid = threadIdx.x, NT = 256;
  const float* x = logits + (long)row * VV;
  __shared__ float red[256];
  float mx = NEG_BIG;
  for (int i = tid; i < VV; i += NT) mx = fmaxf(mx, x[i]);
  red[tid] = mx; __syncthreads();
  for (int s = 128; s > 0; s >>= 1) { if (tid < s) red[tid] = fmaxf(red[tid], red[tid + s]); __syncthreads(); }
  float rmax = red[0]; __syncthreads();
  float sm = 0.0f;
  for (int i = tid; i < VV; i += NT) sm += __expf(x[i] - rmax);
  red[tid] = sm; __syncthreads();
  for (int s = 128; s > 0; s >>= 1) { if (tid < s) red[tid] += red[tid + s]; __syncthreads(); }
  float lse = rmax + logf(red[0]);

  float tv[8]; int ti[8];
#pragma unroll
  for (int q = 0; q < 8; ++q) { tv[q] = NEG_BIG; ti[q] = 0; }
  for (int i = tid; i < VV; i += NT) {
    float v = x[i];
    if (v > tv[7]) {
      int p = 7;
      while (p > 0 && tv[p - 1] < v) { tv[p] = tv[p - 1]; ti[p] = ti[p - 1]; --p; }
      tv[p] = v; ti[p] = i;
    }
  }
  __shared__ float sv[256 * 8];
  __shared__ int   si[256 * 8];
#pragma unroll
  for (int q = 0; q < 8; ++q) { sv[tid * 8 + q] = tv[q]; si[tid * 8 + q] = ti[q]; }
  __syncthreads();
  if (tid == 0) {
    float base = nls[row];
    int   m    = mask[row];
    for (int s = 0; s < 8; ++s) {
      float best = NEG_BIG; int bp = 0;
      for (int i = 0; i < 256 * 8; ++i) {
        if (sv[i] > best || (sv[i] == best && si[i] < si[bp])) { best = sv[i]; bp = i; }
      }
      sv[bp] = NEG_BIG;
      if (m) { cvals[row * 8 + s] = (best - lse) + base; cidx[row * 8 + s] = si[bp]; }
      else   { cvals[row * 8 + s] = base;                cidx[row * 8 + s] = 0; }
    }
  }
}

// Per-batch: combine K*BEAM=64 candidates -> top 8 (stable), emit tokens/preds,
// next-step beam state; pred feeds state gather.
__global__ void combine_kernel(const float* __restrict__ cvals, const int* __restrict__ cidx,
                               const int* __restrict__ mask_cur, int* __restrict__ mask_nxt,
                               float* __restrict__ nls_nxt, int* __restrict__ tok_nxt,
                               int* __restrict__ pred, int* __restrict__ out_sent,
                               int* __restrict__ out_pred, int t) {
  int b = blockIdx.x;
  if (threadIdx.x != 0) return;
  float v[64]; int tk[64];
  for (int c = 0; c < 64; ++c) {
    v[c]  = cvals[(b * 8 + (c >> 3)) * 8 + (c & 7)];   // flat = k*BEAM + j
    tk[c] = cidx [(b * 8 + (c >> 3)) * 8 + (c & 7)];
  }
  for (int s = 0; s < 8; ++s) {
    float best = NEG_BIG; int bc = 0;
    for (int c = 0; c < 64; ++c)
      if (v[c] > best) { best = v[c]; bc = c; }        // strict > == stable tie-break
    v[bc] = NEG_BIG;
    int p = bc >> 3;
    out_sent[(b * TT + t) * KB + s] = tk[bc];
    out_pred[(b * TT + t) * KB + s] = p;
    tok_nxt [b * 8 + s] = tk[bc];
    nls_nxt [b * 8 + s] = best;
    mask_nxt[b * 8 + s] = mask_cur[b * 8 + p];
    pred    [b * 8 + s] = p;
  }
}

// Gather 5 states along beam axis by predecessor (double-buffered)
__global__ void gather_kernel(const float* __restrict__ Scur, float* __restrict__ Snxt,
                              const int* __restrict__ pred) {
  int i = blockIdx.x * blockDim.x + threadIdx.x;      // 64*1024
  int row = i >> 10, j = i & (HH - 1);
  int src = (row & ~7) + pred[row];
#pragma unroll
  for (int st = 0; st < 5; ++st)
    Snxt[st * RR * HH + row * HH + j] = Scur[st * RR * HH + src * HH + j];
}

// ---------------------------------------------------------------------------
extern "C" void kernel_launch(void* const* d_in, const int* in_sizes, int n_in,
                              void* d_out, int out_size, void* d_ws, size_t ws_size,
                              hipStream_t stream) {
  const int*   summ = (const int*)d_in[0];
  const float* enc  = (const float*)d_in[1];
  const float* h1i  = (const float*)d_in[2];
  const float* c1i  = (const float*)d_in[3];
  const float* h2i  = (const float*)d_in[4];
  const float* c2i  = (const float*)d_in[5];
  const float* emb  = (const float*)d_in[6];
  const float* W1   = (const float*)d_in[7];
  const float* b1   = (const float*)d_in[8];
  const float* W2   = (const float*)d_in[9];
  const float* b2   = (const float*)d_in[10];
  const float* Wc   = (const float*)d_in[11];
  const float* Wout = (const float*)d_in[12];
  const float* bout = (const float*)d_in[13];

  char* ws = (char*)d_ws;
  size_t off = 0;
  auto alloc = [&](size_t bytes) -> char* {
    char* p = ws + off;
    off += (bytes + 255) & ~(size_t)255;
    return p;
  };
  // Transposed bf16 weights: Wt[N x K]
  bf16_t* W1t   = (bf16_t*)alloc((size_t)4096 * 2560 * 2);
  bf16_t* W2t   = (bf16_t*)alloc((size_t)4096 * 2048 * 2);
  bf16_t* Wct   = (bf16_t*)alloc((size_t)1024 * 2048 * 2);
  bf16_t* Woutt = (bf16_t*)alloc((size_t)32000 * 1024 * 2);
  bf16_t* encb  = (bf16_t*)alloc((size_t)BB * LL * HH * 2);   // [b][l][h]
  bf16_t* encTb = (bf16_t*)alloc((size_t)BB * HH * LL * 2);   // [b][h][l]
  float*  S0    = (float*) alloc((size_t)5 * RR * HH * 4);
  float*  S1    = (float*) alloc((size_t)5 * RR * HH * 4);
  bf16_t* X1b   = (bf16_t*)alloc((size_t)RR * 2560 * 2);
  bf16_t* X2b   = (bf16_t*)alloc((size_t)RR * 2048 * 2);
  bf16_t* XCb   = (bf16_t*)alloc((size_t)RR * 2048 * 2);
  bf16_t* h2b   = (bf16_t*)alloc((size_t)RR * HH * 2);
  bf16_t* hattb = (bf16_t*)alloc((size_t)RR * HH * 2);
  bf16_t* attb  = (bf16_t*)alloc((size_t)RR * LL * 2);
  float*  Z     = (float*) alloc((size_t)RR * 4096 * 4);
  float*  scoresF = (float*)alloc((size_t)RR * LL * 4);
  float*  logits  = (float*)alloc((size_t)RR * VV * 4);
  float*  cvals   = (float*)alloc((size_t)RR * 8 * 4);
  int*    cidx    = (int*)  alloc((size_t)RR * 8 * 4);
  int*    tokA    = (int*)  alloc(RR * 4);
  int*    tokB    = (int*)  alloc(RR * 4);
  int*    maskA   = (int*)  alloc(RR * 4);
  int*    maskB   = (int*)  alloc(RR * 4);
  float*  nlsA    = (float*)alloc(RR * 4);
  float*  nlsB    = (float*)alloc(RR * 4);
  int*    pred    = (int*)  alloc(RR * 4);

  int* out_sent = (int*)d_out;                  // [B, T, BEAM]
  int* out_pred = out_sent + BB * TT * KB;      // [B, T, BEAM]

  // --- one-time conversions: transpose + bf16 ---
  // W [R x C] -> Wt [C x R]
  f2bfT_kernel<<<dim3(4096 / 32, 2560 / 32, 1), 256, 0, stream>>>(W1, W1t, 2560, 4096, 0, 0);
  f2bfT_kernel<<<dim3(4096 / 32, 2048 / 32, 1), 256, 0, stream>>>(W2, W2t, 2048, 4096, 0, 0);
  f2bfT_kernel<<<dim3(1024 / 32, 2048 / 32, 1), 256, 0, stream>>>(Wc, Wct, 2048, 1024, 0, 0);
  f2bfT_kernel<<<dim3(32000 / 32, 1024 / 32, 1), 256, 0, stream>>>(Wout, Woutt, 1024, 32000, 0, 0);
  // enc [b][L x H] -> encT [b][H x L]  (batched via grid.z)
  f2bfT_kernel<<<dim3(HH / 32, LL / 32, BB), 256, 0, stream>>>(enc, encTb, LL, HH,
                                                               (long)LL * HH, (long)HH * LL);
  {
    long n = (long)BB * LL * HH;
    f2bf_kernel<<<(int)((n + 255) / 256), 256, 0, stream>>>(enc, encb, n);
  }

  init_kernel<<<RR * HH / 256, 256, 0, stream>>>(summ, h1i, c1i, h2i, c2i,
                                                 S0, tokA, maskA, nlsA);

  for (int t = 0; t < TT; ++t) {
    const bool odd = (t & 1) != 0;
    float* Sc = odd ? S1 : S0;    float* Sn = odd ? S0 : S1;
    int*   tc = odd ? tokB : tokA;  int*   tn = odd ? tokA : tokB;
    int*   mc = odd ? maskB : maskA; int*  mn = odd ? maskA : maskB;
    float* lc = odd ? nlsB : nlsA;  float* ln = odd ? nlsA : nlsB;

    mask_kernel<<<1, 64, 0, stream>>>(tc, mc);
    build_x1_kernel<<<RR * 2560 / 256, 256, 0, stream>>>(tc, emb, Sc, X1b);

    // LSTM1: [64,2560] @ [2560,4096]
    gemm_kernel<<<dim3(32, 4), 256, 0, stream>>>(X1b, 2560, W1t, 2560, b1, Z, 4096, 4096, 2560);
    lstm1_pw_kernel<<<RR * HH / 256, 256, 0, stream>>>(Z, Sc, X2b);

    // LSTM2: [64,2048] @ [2048,4096]
    gemm_kernel<<<dim3(32, 4), 256, 0, stream>>>(X2b, 2048, W2t, 2048, b2, Z, 4096, 4096, 2048);
    lstm2_pw_kernel<<<RR * HH / 256, 256, 0, stream>>>(Z, Sc, h2b, XCb);

    // attention
    scores_kernel<<<dim3(2, BB), 256, 0, stream>>>(h2b, encb, scoresF);
    softmax_kernel<<<RR, 256, 0, stream>>>(scoresF, attb);
    ctx_kernel<<<dim3(8, BB), 256, 0, stream>>>(attb, encTb, XCb);

    // hatt = tanh([h2|ctx] @ Wc): [64,2048] @ [2048,1024]
    gemm_kernel<<<dim3(8, 4), 256, 0, stream>>>(XCb, 2048, Wct, 2048, nullptr, Z, 1024, 1024, 2048);
    tanh_pw_kernel<<<RR * HH / 256, 256, 0, stream>>>(Z, Sc, hattb);

    // vocab projection (dominant): [64,1024] @ [1024,32000]
    gemm_kernel<<<dim3(250, 4), 256, 0, stream>>>(hattb, 1024, Woutt, 1024, bout,
                                                  logits, 32000, 32000, 1024);

    topk_kernel<<<RR, 256, 0, stream>>>(logits, mc, lc, cvals, cidx);
    combine_kernel<<<BB, 32, 0, stream>>>(cvals, cidx, mc, mn, ln, tn, pred,
                                          out_sent, out_pred, t);
    gather_kernel<<<RR * HH / 256, 256, 0, stream>>>(Sc, Sn, pred);
  }
  (void)in_sizes; (void)n_in; (void)out_size; (void)ws_size;
}
